// IsotropicSkillCodec_64029372449368
// MI455X (gfx1250) — compile-verified, matrix operand order checked by reading, not run on hardware
//
#include <hip/hip_runtime.h>
#include <hip/hip_bf16.h>

typedef __attribute__((ext_vector_type(16))) __bf16 v16bf;
typedef __attribute__((ext_vector_type(8)))  float  v8f;

#define BATCH      32768
#define EMBED      1024
#define NUM_TOK    32
#define TOK_DIM    32
#define CBSZ       256
#define ROWS_WG    64
#define MTILES     (ROWS_WG / 16)        // 4
#define WAVES_WG   4
#define THREADS_WG 128
#define NUM_WG     (BATCH / ROWS_WG)     // 512
#define KCHUNKS    (EMBED / 32)          // 32
#define NTILES     (EMBED / 16)          // 64

// ---- workspace layout (bytes) ----
#define WS_ENCB    0                       // bf16 [64][32][32][16]  = 2 MB
#define WS_DECB    (2u*1024*1024)          // bf16                  = 2 MB
#define WS_CBB     (4u*1024*1024)          // bf16 [16][32][16]     = 16 KB
#define WS_CBN     (WS_CBB + 16*1024)      // f32 [256]             = 1 KB
#define WS_ACC     (WS_CBN + 1024)         // f32 vq_sum, recon_sum

// ---- encoder LDS layout (bytes) ----
#define LDS_AT     0                       // bf16 [64][1024] = 131072
#define LDS_CBN    131072                  // f32 [256]       = 1024
#define LDS_CBB    132096                  // bf16 [16][32][16] = 16384
#define LDS_STG    148480                  // bf16 [4][16][32]  = 4096
#define ENC_LDS    152576
#define DEC_LDS    131072

// ---- output layout (elements) ----
#define OUT_REC_N   (BATCH * EMBED)
#define OUT_CODES_N (BATCH * NUM_TOK)

union FragB { v16bf v; uint4 q[2]; };

__device__ __forceinline__ v8f wmma_bf16(const FragB& a, const FragB& b, v8f c) {
    return __builtin_amdgcn_wmma_f32_16x16x32_bf16(false, a.v, false, b.v,
                                                   (short)0, c, false, false);
}

// Fragment swizzle (per ISA 7.12.2, 16-bit A/B layout):
// lane<16 holds K = {khi+0..7, 16+khi+0..7} with khi = (lane>=16)*8.
__device__ __forceinline__ int frag_k(int lane, int j) {
    int khi = ((lane >> 4) & 1) * 8;
    return (j < 8) ? (khi + j) : (16 + khi + (j - 8));
}

// ---------------------------------------------------------------------------
// prep: swizzle weights to bf16 WMMA-B fragments, bake -2*codebook, norms.
// ---------------------------------------------------------------------------
__global__ void vq_prep_kernel(const float* __restrict__ enc_w,
                               const float* __restrict__ dec_w,
                               const float* __restrict__ codebook,
                               char* __restrict__ ws) {
    int g = blockIdx.x * blockDim.x + threadIdx.x;   // 0 .. 1M-1
    __bf16* encB = (__bf16*)(ws + WS_ENCB);
    __bf16* decB = (__bf16*)(ws + WS_DECB);
    __bf16* cbB  = (__bf16*)(ws + WS_CBB);
    float*  cbn  = (float*)(ws + WS_CBN);
    float*  acc  = (float*)(ws + WS_ACC);

    int j    = g & 15;
    int lane = (g >> 4) & 31;
    int kc   = (g >> 9) & 31;
    int nt   = g >> 14;                               // 0..63
    int k    = kc * 32 + frag_k(lane, j);
    int n    = nt * 16 + (lane & 15);
    encB[g] = (__bf16)enc_w[n * EMBED + k];
    decB[g] = (__bf16)dec_w[n * EMBED + k];

    if (g < 16 * 32 * 16) {                           // codebook B fragments
        int cj = g & 15, cl = (g >> 4) & 31, e = g >> 9;   // e: entry tile 0..15
        int ck = frag_k(cl, cj);                       // 0..31 (single K chunk)
        int cn = e * 16 + (cl & 15);                   // entry index
        cbB[g] = (__bf16)(-2.0f * codebook[cn * TOK_DIM + ck]);
    }
    if (g < CBSZ) {
        float s = 0.f;
        #pragma unroll
        for (int d = 0; d < TOK_DIM; ++d) { float v = codebook[g * TOK_DIM + d]; s += v * v; }
        cbn[g] = s;
    }
    if (g == 0) { acc[0] = 0.f; acc[1] = 0.f; }
}

// ---------------------------------------------------------------------------
// fused encoder GEMM + VQ argmin (projected never hits HBM).
// Waves split the token (N) dimension; each wave covers all 4 M-tiles so a
// B fragment is loaded once per WG and reused by 4 WMMAs (4x less L2 traffic).
// ---------------------------------------------------------------------------
__global__ void vq_encode_kernel(const float* __restrict__ skills,
                                 const float* __restrict__ enc_b,
                                 const char*  __restrict__ ws,
                                 int* __restrict__ out_codes,
                                 float* __restrict__ vq_accum) {
    extern __shared__ __align__(16) char smem[];
    __bf16* At   = (__bf16*)(smem + LDS_AT);
    float*  cbnS = (float*)(smem + LDS_CBN);
    __bf16* cbBS = (__bf16*)(smem + LDS_CBB);
    const __bf16* encB = (const __bf16*)(ws + WS_ENCB);
    const __bf16* cbBG = (const __bf16*)(ws + WS_CBB);
    const float*  cbnG = (const float*)(ws + WS_CBN);

    const int tid  = threadIdx.x;
    const int m0   = blockIdx.x * ROWS_WG;
    const int wave = tid >> 5, lane = tid & 31;
    const int l15  = lane & 15, half = lane >> 4;          // 0/1
    const int khi  = half * 8;

    // stage skills tile -> bf16 LDS; codebook fragments + norms -> LDS
    for (int i = tid; i < 256; i += THREADS_WG) cbnS[i] = cbnG[i];
    for (int i = tid; i < 16 * 32 * 16 / 8; i += THREADS_WG)
        ((uint4*)cbBS)[i] = ((const uint4*)cbBG)[i];
    for (int idx = tid; idx < ROWS_WG * (EMBED / 4); idx += THREADS_WG) {
        int row = idx >> 8, c4 = (idx & 255) * 4;
        float4 f = *(const float4*)(skills + (size_t)(m0 + row) * EMBED + c4);
        __bf16* p = At + row * EMBED + c4;
        p[0] = (__bf16)f.x; p[1] = (__bf16)f.y; p[2] = (__bf16)f.z; p[3] = (__bf16)f.w;
    }
    __syncthreads();

    __bf16* stg = (__bf16*)(smem + LDS_STG) + wave * (16 * 32);
    float local_vq = 0.f;

    for (int t = wave; t < NUM_TOK; t += WAVES_WG) {
        v8f acc[MTILES][2];
        #pragma unroll
        for (int m = 0; m < MTILES; ++m) { acc[m][0] = (v8f){}; acc[m][1] = (v8f){}; }

        const __bf16* pB0 = encB + ((size_t)((2 * t) * 32) * 32 + lane) * 16;
        const __bf16* pB1 = encB + ((size_t)((2 * t + 1) * 32) * 32 + lane) * 16;
        for (int kc = 0; kc < KCHUNKS; ++kc) {
            FragB b0, b1;
            b0.v = *(const v16bf*)(pB0 + (size_t)kc * 512);
            b1.v = *(const v16bf*)(pB1 + (size_t)kc * 512);
            __builtin_prefetch(pB0 + (size_t)(kc + 1) * 512, 0, 1);
            __builtin_prefetch(pB1 + (size_t)(kc + 1) * 512, 0, 1);
            #pragma unroll
            for (int m = 0; m < MTILES; ++m) {
                FragB a;
                const __bf16* ap = At + (m * 16 + l15) * EMBED + kc * 32 + khi;
                a.q[0] = *(const uint4*)ap;
                a.q[1] = *(const uint4*)(ap + 16);
                acc[m][0] = wmma_bf16(a, b0, acc[m][0]);
                acc[m][1] = wmma_bf16(a, b1, acc[m][1]);
            }
        }
        const float bia0 = enc_b[t * TOK_DIM + l15];
        const float bia1 = enc_b[t * TOK_DIM + 16 + l15];

        #pragma unroll
        for (int m = 0; m < MTILES; ++m) {
            // bias, stage fp32 block as bf16 [16 rows][32 dims], ||x||^2 partials
            float xn[8];
            #pragma unroll
            for (int i = 0; i < 8; ++i) {
                float a0 = acc[m][0][i] + bia0;
                float a1 = acc[m][1][i] + bia1;
                int r = i + half * 8;                      // C layout: lanes>=16 hold M=8..15
                stg[r * 32 + l15]      = (__bf16)a0;
                stg[r * 32 + 16 + l15] = (__bf16)a1;
                xn[i] = a0 * a0 + a1 * a1;
            }
            // re-fragment the token block as WMMA-A from LDS
            FragB ap;
            ap.q[0] = *(const uint4*)(stg + l15 * 32 + khi);
            ap.q[1] = *(const uint4*)(stg + l15 * 32 + 16 + khi);

            // distance GEMM: 16 tiles of 16 entries, rel = -2 x.c + ||c||^2
            float bestv[8]; int besti[8];
            #pragma unroll
            for (int i = 0; i < 8; ++i) { bestv[i] = 3.0e38f; besti[i] = 0; }
            for (int e = 0; e < 16; ++e) {
                FragB cb;
                const __bf16* cp = cbBS + ((size_t)e * 32 + lane) * 16;
                cb.q[0] = *(const uint4*)cp;
                cb.q[1] = *(const uint4*)(cp + 8);
                v8f z = {};
                v8f d = wmma_bf16(ap, cb, z);
                float cn = cbnS[e * 16 + l15];
                #pragma unroll
                for (int i = 0; i < 8; ++i) {
                    float val = d[i] + cn;
                    if (val < bestv[i]) { bestv[i] = val; besti[i] = e * 16 + l15; }
                }
            }
            // butterfly argmin + ||x||^2 sum across lanes 0..15 (halves indep.)
            #pragma unroll
            for (int off = 8; off >= 1; off >>= 1) {
                #pragma unroll
                for (int i = 0; i < 8; ++i) {
                    float ov = __shfl_xor(bestv[i], off, 32);
                    int   oi = __shfl_xor(besti[i], off, 32);
                    float ox = __shfl_xor(xn[i],    off, 32);
                    xn[i] += ox;
                    if (ov < bestv[i]) { bestv[i] = ov; besti[i] = oi; }
                }
            }
            if (l15 == 0) {
                int rbase = m0 + m * 16 + half * 8;
                #pragma unroll
                for (int i = 0; i < 8; ++i) {
                    out_codes[(size_t)(rbase + i) * NUM_TOK + t] = besti[i];
                    local_vq += xn[i] + bestv[i];          // ||x-c||^2
                }
            }
        }
    }
    if (l15 == 0) atomicAdd(vq_accum, local_vq);
}

// ---------------------------------------------------------------------------
// decoder GEMM from codes (st == quantized in forward) + recon loss.
// Waves split the N (embed) dimension; B fragment reused across 4 M-tiles.
// ---------------------------------------------------------------------------
__global__ void vq_decode_kernel(const int* __restrict__ codes,
                                 const float* __restrict__ codebook,
                                 const float* __restrict__ dec_b,
                                 const float* __restrict__ skills,
                                 const char* __restrict__ ws,
                                 float* __restrict__ out_rec,
                                 float* __restrict__ rec_accum) {
    extern __shared__ __align__(16) char smem[];
    __bf16* Qt = (__bf16*)smem;
    const __bf16* decB = (const __bf16*)(ws + WS_DECB);

    const int tid  = threadIdx.x;
    const int m0   = blockIdx.x * ROWS_WG;
    const int wave = tid >> 5, lane = tid & 31;
    const int l15  = lane & 15, half = lane >> 4;
    const int khi  = half * 8;

    // gather quantized tile: Qt[row][t*32+d] = codebook[codes[row][t]][d]
    for (int p = tid; p < ROWS_WG * NUM_TOK; p += THREADS_WG) {
        int row = p >> 5, t = p & 31;
        int code = codes[(size_t)(m0 + row) * NUM_TOK + t];
        const float* cb = codebook + (size_t)code * TOK_DIM;
        __bf16* q = Qt + row * EMBED + t * TOK_DIM;
        #pragma unroll
        for (int d = 0; d < TOK_DIM; ++d) q[d] = (__bf16)cb[d];
    }
    __syncthreads();

    float local_r = 0.f;
    for (int nt = wave; nt < NTILES; nt += WAVES_WG) {
        v8f acc[MTILES];
        #pragma unroll
        for (int m = 0; m < MTILES; ++m) acc[m] = (v8f){};

        const __bf16* pB = decB + ((size_t)(nt * 32) * 32 + lane) * 16;
        for (int kc = 0; kc < KCHUNKS; ++kc) {
            FragB b;
            b.v = *(const v16bf*)(pB + (size_t)kc * 512);
            __builtin_prefetch(pB + (size_t)(kc + 1) * 512, 0, 1);
            #pragma unroll
            for (int m = 0; m < MTILES; ++m) {
                FragB a;
                const __bf16* ap = Qt + (m * 16 + l15) * EMBED + kc * 32 + khi;
                a.q[0] = *(const uint4*)ap;
                a.q[1] = *(const uint4*)(ap + 16);
                acc[m] = wmma_bf16(a, b, acc[m]);
            }
        }
        int col = nt * 16 + l15;
        float bias = dec_b[col];
        #pragma unroll
        for (int m = 0; m < MTILES; ++m) {
            #pragma unroll
            for (int i = 0; i < 8; ++i) {
                int row = m0 + m * 16 + half * 8 + i;
                float v = acc[m][i] + bias;
                __builtin_nontemporal_store(v, out_rec + (size_t)row * EMBED + col);
                float s = __builtin_nontemporal_load(skills + (size_t)row * EMBED + col);
                float diff = v - s;
                local_r += diff * diff;
            }
        }
    }
    atomicAdd(rec_accum, local_r);
}

__global__ void vq_finalize_kernel(const char* __restrict__ ws,
                                   float* __restrict__ out_loss) {
    const float* acc = (const float*)(ws + WS_ACC);
    const float inv = 1.0f / ((float)BATCH * (float)EMBED);
    // vq_loss = codebook_loss + 0.25*commitment = 1.25 * mean(||p-q||^2)
    out_loss[0] = 1.25f * acc[0] * inv + acc[1] * inv;
}

extern "C" void kernel_launch(void* const* d_in, const int* in_sizes, int n_in,
                              void* d_out, int out_size, void* d_ws, size_t ws_size,
                              hipStream_t stream) {
    const float* skills   = (const float*)d_in[0];
    const float* enc_w    = (const float*)d_in[1];
    const float* enc_b    = (const float*)d_in[2];
    const float* codebook = (const float*)d_in[3];
    const float* dec_w    = (const float*)d_in[4];
    const float* dec_b    = (const float*)d_in[5];

    char*  ws        = (char*)d_ws;
    float* out_rec   = (float*)d_out;
    int*   out_codes = (int*)(out_rec + OUT_REC_N);
    float* out_loss  = out_rec + OUT_REC_N + OUT_CODES_N;
    float* vq_acc    = (float*)(ws + WS_ACC);
    float* rec_acc   = vq_acc + 1;

    hipFuncSetAttribute((const void*)vq_encode_kernel,
                        hipFuncAttributeMaxDynamicSharedMemorySize, ENC_LDS);
    hipFuncSetAttribute((const void*)vq_decode_kernel,
                        hipFuncAttributeMaxDynamicSharedMemorySize, DEC_LDS);

    vq_prep_kernel<<<4096, 256, 0, stream>>>(enc_w, dec_w, codebook, ws);
    vq_encode_kernel<<<NUM_WG, THREADS_WG, ENC_LDS, stream>>>(
        skills, enc_b, ws, out_codes, vq_acc);
    vq_decode_kernel<<<NUM_WG, THREADS_WG, DEC_LDS, stream>>>(
        out_codes, codebook, dec_b, skills, ws, out_rec, rec_acc);
    vq_finalize_kernel<<<1, 1, 0, stream>>>(ws, out_loss);
}
